// TNTDecoder_79697413145190
// MI455X (gfx1250) — compile-verified, matrix operand order checked by reading, not run on hardware
//
#include <hip/hip_runtime.h>
#include <hip/hip_bf16.h>
#include <math.h>

// Problem constants (match reference)
#define B_   256
#define N_   2000
#define M_   50
#define C_   128
#define H_   64
#define TD_  60      // HORIZON*2
#define LN_EPS 1e-5f

typedef __attribute__((ext_vector_type(16))) _Float16 v16h;
typedef __attribute__((ext_vector_type(8)))  float    v8f;

union HFrag { v16h v; _Float16 h[16]; };

struct MlpP {
  const float *W1, *b1, *g1, *e1, *W2, *b2, *g2, *e2, *W3, *b3;
};

__device__ __forceinline__ float xr(float v, int m) { return __shfl_xor(v, m, 32); }

// Wave-local LDS RAW fence (CDNA5 split wait counters)
__device__ __forceinline__ void lds_fence_wave() {
  asm volatile("s_wait_dscnt 0" ::: "memory");
}

// Assemble a 16-half WMMA fragment from two contiguous 16-byte LDS slices.
__device__ __forceinline__ v16h load_frag(const _Float16* p0, const _Float16* p1) {
  HFrag f;
  *reinterpret_cast<uint4*>(&f.h[0]) = *reinterpret_cast<const uint4*>(p0);
  *reinterpret_cast<uint4*>(&f.h[8]) = *reinterpret_cast<const uint4*>(p1);
  return f.v;
}

// Load the 8 loop-invariant W2 B-fragments (64x64, f16, col-major in LDS)
// into registers: bf[ct][kt], ct = 16-col tile, kt = 32-K tile.
__device__ __forceinline__ void load_w2_frags(const _Float16* w2t, v16h bf[4][2]) {
  const int lane = threadIdx.x & 31;
  const int ko   = (lane >> 4) * 8;
  const int m16  = lane & 15;
#pragma unroll
  for (int ct = 0; ct < 4; ++ct) {
    const _Float16* wc = w2t + (ct * 16 + m16) * H_;
    bf[ct][0] = load_frag(wc + ko,      wc + 16 + ko);    // K 0..31
    bf[ct][1] = load_frag(wc + 32 + ko, wc + 48 + ko);    // K 32..63
  }
}

// Hidden layer for one 16-row tile:
//   h2n = relu(LN(aTile(16x64,f16) @ W2(64x64) + b2) * g2 + e2)
// aT: row-major 16x64 f16 (per-wave LDS); bf: W2 fragments in registers.
// b2v/g2v/e2v: per-lane column params (col = ct*16 + lane&15), preloaded.
// Output: per-lane C-layout h2n[ct][v]; row = v + 8*(lane>=16), col = ct*16+(lane&15).
__device__ __forceinline__ void hidden_wmma_regs(const _Float16* aT, const v16h bf[4][2],
                                                 const float b2v[4], const float g2v[4],
                                                 const float e2v[4], float h2n[4][8])
{
  const int lane = threadIdx.x & 31;
  const int m16  = lane & 15;
  const int ko   = (lane >> 4) * 8;

  // A fragments: lane m16 holds row m16; K split per 16-lane half (ISA 7.12.2)
  v16h aLo = load_frag(aT + m16 * H_ + ko,      aT + m16 * H_ + 16 + ko);   // K 0..31
  v16h aHi = load_frag(aT + m16 * H_ + 32 + ko, aT + m16 * H_ + 48 + ko);   // K 32..63

  v8f acc[4];
#pragma unroll
  for (int ct = 0; ct < 4; ++ct) {
    v8f c = {};
    c = __builtin_amdgcn_wmma_f32_16x16x32_f16(false, aLo, false, bf[ct][0], (short)0, c, false, false);
    c = __builtin_amdgcn_wmma_f32_16x16x32_f16(false, aHi, false, bf[ct][1], (short)0, c, false, false);
    acc[ct] = c;
  }

  // LayerNorm over 64 cols per row: 4 regs (ct) per lane x 16 lanes per half.
#pragma unroll
  for (int v = 0; v < 8; ++v) {
    float s = 0.f, s2 = 0.f;
    float x[4];
#pragma unroll
    for (int ct = 0; ct < 4; ++ct) {
      float t = acc[ct][v] + b2v[ct];
      x[ct] = t; s += t; s2 += t * t;
    }
    for (int mk = 1; mk < 16; mk <<= 1) { s += xr(s, mk); s2 += xr(s2, mk); }
    float mu  = s * (1.f / 64.f);
    float var = s2 * (1.f / 64.f) - mu * mu;
    float r   = rsqrtf(var + LN_EPS);
#pragma unroll
    for (int ct = 0; ct < 4; ++ct) {
      float y = (x[ct] - mu) * r * g2v[ct] + e2v[ct];
      h2n[ct][v] = fmaxf(y, 0.f);
    }
  }
}

// ---------------------------------------------------------------------------
// Kernel 1: per-batch fold of the constant feature through W1[0:128,:] of all
// four MLPs: featWS[w][b][j] = sum_k feat[b,k] * W1w[k,j].  Tiny (8.4M MACs).
// ---------------------------------------------------------------------------
__global__ void feat_proj_kernel(const float* __restrict__ feat,
                                 const float* __restrict__ W1p, const float* __restrict__ W1o,
                                 const float* __restrict__ W1m, const float* __restrict__ W1s,
                                 float* __restrict__ outWS)
{
  const int b = blockIdx.x, w = blockIdx.y, j = threadIdx.x;
  const float* W = (w == 0) ? W1p : (w == 1) ? W1o : (w == 2) ? W1m : W1s;
  const float* f = feat + b * C_;
  float s = 0.f;
  for (int k = 0; k < C_; ++k) s += f[k] * W[k * H_ + j];
  outWS[(w * B_ + b) * H_ + j] = s;
}

// ---------------------------------------------------------------------------
// Kernel 2: TargetPred — prob (dout=1 -> logits+mask) and offset (dout=2).
// grid (B, 2); 4 waves/block; each wave owns a 16-row tile.  W2 fragments and
// all per-lane layer-2/3 params are hoisted into registers outside the loop.
// ---------------------------------------------------------------------------
__global__ void __launch_bounds__(128)
stage1_kernel(MlpP prob, MlpP offs,
              const float* __restrict__ featWS, const float* __restrict__ cand,
              const float* __restrict__ mask,
              float* __restrict__ logits, float* __restrict__ offsets)
{
  const int b     = blockIdx.x;
  const int which = blockIdx.y;
  const MlpP mp   = (which == 0) ? prob : offs;
  const int dout  = (which == 0) ? 1 : 2;
  const float* featX = featWS + (which == 0 ? 0 : B_ * H_) + b * H_;

  __shared__ __align__(16) _Float16 sW2t[H_ * H_];
  __shared__ __align__(16) _Float16 sA[4][16 * H_];
  __shared__ float sFeat[H_], sW1c0[H_], sW1c1[H_];
  __shared__ float sB1[H_], sG1[H_], sE1[H_], sB2[H_], sG2[H_], sE2[H_];
  __shared__ float sW3[H_ * 2];
  __shared__ float sB3[2];

  const int tid = threadIdx.x;
  for (int i = tid; i < H_ * H_; i += 128) {
    int col = i >> 6, k = i & 63;
    sW2t[i] = (_Float16)mp.W2[k * H_ + col];
  }
  for (int i = tid; i < H_; i += 128) {
    sFeat[i]  = featX[i];
    sW1c0[i]  = mp.W1[C_ * H_ + i];         // W1 row 128 (cand.x)
    sW1c1[i]  = mp.W1[(C_ + 1) * H_ + i];   // W1 row 129 (cand.y)
    sB1[i] = mp.b1[i]; sG1[i] = mp.g1[i]; sE1[i] = mp.e1[i];
    sB2[i] = mp.b2[i]; sG2[i] = mp.g2[i]; sE2[i] = mp.e2[i];
  }
  for (int i = tid; i < H_ * dout; i += 128) sW3[i] = mp.W3[i];
  if (tid < dout) sB3[tid] = mp.b3[tid];
  __syncthreads();

  const int wave = tid >> 5, lane = tid & 31;
  const int half = lane >> 4, m16 = lane & 15;
  _Float16* aT = sA[wave];

  // Loop-invariant register state
  v16h bf[4][2];
  load_w2_frags(sW2t, bf);
  float b2v[4], g2v[4], e2v[4], w30[4], w31[4];
#pragma unroll
  for (int ct = 0; ct < 4; ++ct) {
    int col = ct * 16 + m16;
    b2v[ct] = sB2[col]; g2v[ct] = sG2[col]; e2v[ct] = sE2[col];
    w30[ct] = sW3[col * dout + 0];
    w31[ct] = (dout == 2) ? sW3[col * dout + 1] : 0.f;
  }

  for (int t = wave; t < N_ / 16; t += 4) {
    const int rbase = t * 16;
    const int row   = rbase + (lane >> 1);
    const float2 cc = reinterpret_cast<const float2*>(cand)[b * N_ + row];
    if (t + 4 < N_ / 16)
      __builtin_prefetch(&cand[(b * N_ + row + 64) * 2], 0, 1);   // global_prefetch_b8

    // Layer 1 + LN + ReLU -> f16 A-tile (2 lanes per row, 32 cols each)
    {
      const int cb = (lane & 1) * 32;
      float h[32]; float s = 0.f, s2 = 0.f;
#pragma unroll
      for (int j = 0; j < 32; ++j) {
        int c = cb + j;
        float x = sFeat[c] + cc.x * sW1c0[c] + cc.y * sW1c1[c] + sB1[c];
        h[j] = x; s += x; s2 += x * x;
      }
      s += xr(s, 1); s2 += xr(s2, 1);
      float mu  = s * (1.f / 64.f);
      float var = s2 * (1.f / 64.f) - mu * mu;
      float r   = rsqrtf(var + LN_EPS);
      _Float16* aRow = aT + (lane >> 1) * H_;
#pragma unroll
      for (int j = 0; j < 32; ++j) {
        int c = cb + j;
        float y = (h[j] - mu) * r * sG1[c] + sE1[c];
        aRow[c] = (_Float16)fmaxf(y, 0.f);
      }
    }
    lds_fence_wave();

    float h2n[4][8];
    hidden_wmma_regs(aT, bf, b2v, g2v, e2v, h2n);

    // Layer 3 (dout = 1 or 2) via in-wave reduction
#pragma unroll
    for (int v = 0; v < 8; ++v) {
      float p0 = 0.f, p1 = 0.f;
#pragma unroll
      for (int ct = 0; ct < 4; ++ct) {
        p0 += h2n[ct][v] * w30[ct];
        if (dout == 2) p1 += h2n[ct][v] * w31[ct];
      }
      for (int mk = 1; mk < 16; mk <<= 1) {
        p0 += xr(p0, mk);
        if (dout == 2) p1 += xr(p1, mk);
      }
      if (m16 == 0) {
        int r = rbase + v + half * 8;
        if (dout == 1) {
          logits[b * N_ + r] = p0 + sB3[0] + mask[b * N_ + r];
        } else {
          offsets[(b * N_ + r) * 2 + 0] = p0 + sB3[0];
          offsets[(b * N_ + r) * 2 + 1] = p1 + sB3[1];
        }
      }
    }
  }
}

// ---------------------------------------------------------------------------
// Kernel 3: top-50 of (logits+mask) per batch (softmax is monotone; prob_sel
// is discarded by the reference).  Iterative argmax; destroys ws logits
// (rewritten by stage1 every call -> deterministic).
// ---------------------------------------------------------------------------
__global__ void __launch_bounds__(256)
topk_kernel(float* __restrict__ logits, const float* __restrict__ cand,
            const float* __restrict__ offs, float* __restrict__ tarSel)
{
  const int b = blockIdx.x, tid = threadIdx.x;
  __shared__ float sv[256];
  __shared__ int   si[256];
  float* L = logits + b * N_;

  for (int m = 0; m < M_; ++m) {
    float best = -INFINITY; int bi = N_;
    for (int n = tid; n < N_; n += 256) {
      float v = L[n];
      if (v > best) { best = v; bi = n; }
    }
    sv[tid] = best; si[tid] = bi;
    __syncthreads();
    for (int st = 128; st > 0; st >>= 1) {
      if (tid < st) {
        if (sv[tid + st] > sv[tid] ||
            (sv[tid + st] == sv[tid] && si[tid + st] < si[tid])) {
          sv[tid] = sv[tid + st]; si[tid] = si[tid + st];
        }
      }
      __syncthreads();
    }
    if (tid == 0) {
      int n = si[0];
      L[n] = -INFINITY;
      tarSel[(b * M_ + m) * 2 + 0] = cand[(b * N_ + n) * 2 + 0] + offs[(b * N_ + n) * 2 + 0];
      tarSel[(b * M_ + m) * 2 + 1] = cand[(b * N_ + n) * 2 + 1] + offs[(b * N_ + n) * 2 + 1];
    }
    __syncthreads();
  }
}

// ---------------------------------------------------------------------------
// Kernel 4: MotionEstimation — one block per batch (50 rows in 4 wave-tiles),
// WMMA hidden layer, then 50x60 layer-3 from LDS.
// ---------------------------------------------------------------------------
__global__ void __launch_bounds__(128)
motion_kernel(MlpP mot, const float* __restrict__ featWS,
              const float* __restrict__ tarSel, float* __restrict__ traj_out)
{
  const int b = blockIdx.x;
  __shared__ __align__(16) _Float16 sW2t[H_ * H_];
  __shared__ __align__(16) _Float16 sA[4][16 * H_];
  __shared__ float sH[64][H_];
  __shared__ float sFeat[H_], sW1c0[H_], sW1c1[H_];
  __shared__ float sB1[H_], sG1[H_], sE1[H_], sB2[H_], sG2[H_], sE2[H_];
  __shared__ float sW3[H_ * TD_];
  __shared__ float sB3[TD_];

  const int tid = threadIdx.x;
  for (int i = tid; i < H_ * H_; i += 128) {
    int col = i >> 6, k = i & 63;
    sW2t[i] = (_Float16)mot.W2[k * H_ + col];
  }
  for (int i = tid; i < H_; i += 128) {
    sFeat[i] = featWS[(2 * B_ + b) * H_ + i];
    sW1c0[i] = mot.W1[C_ * H_ + i];
    sW1c1[i] = mot.W1[(C_ + 1) * H_ + i];
    sB1[i] = mot.b1[i]; sG1[i] = mot.g1[i]; sE1[i] = mot.e1[i];
    sB2[i] = mot.b2[i]; sG2[i] = mot.g2[i]; sE2[i] = mot.e2[i];
  }
  for (int i = tid; i < H_ * TD_; i += 128) sW3[i] = mot.W3[i];
  for (int i = tid; i < TD_; i += 128) sB3[i] = mot.b3[i];
  __syncthreads();

  const int wave = tid >> 5, lane = tid & 31;
  const int half = lane >> 4, m16 = lane & 15;
  _Float16* aT = sA[wave];
  const int rbase = wave * 16;
  const int row   = rbase + (lane >> 1);
  const int rl    = (row < M_) ? row : (M_ - 1);
  const float2 cc = reinterpret_cast<const float2*>(tarSel)[b * M_ + rl];

  v16h bf[4][2];
  load_w2_frags(sW2t, bf);
  float b2v[4], g2v[4], e2v[4];
#pragma unroll
  for (int ct = 0; ct < 4; ++ct) {
    int col = ct * 16 + m16;
    b2v[ct] = sB2[col]; g2v[ct] = sG2[col]; e2v[ct] = sE2[col];
  }

  {
    const int cb = (lane & 1) * 32;
    float h[32]; float s = 0.f, s2 = 0.f;
#pragma unroll
    for (int j = 0; j < 32; ++j) {
      int c = cb + j;
      float x = sFeat[c] + cc.x * sW1c0[c] + cc.y * sW1c1[c] + sB1[c];
      h[j] = x; s += x; s2 += x * x;
    }
    s += xr(s, 1); s2 += xr(s2, 1);
    float mu  = s * (1.f / 64.f);
    float var = s2 * (1.f / 64.f) - mu * mu;
    float r   = rsqrtf(var + LN_EPS);
    _Float16* aRow = aT + (lane >> 1) * H_;
#pragma unroll
    for (int j = 0; j < 32; ++j) {
      int c = cb + j;
      float y = (h[j] - mu) * r * sG1[c] + sE1[c];
      aRow[c] = (_Float16)fmaxf(y, 0.f);
    }
  }
  lds_fence_wave();

  float h2n[4][8];
  hidden_wmma_regs(aT, bf, b2v, g2v, e2v, h2n);

#pragma unroll
  for (int ct = 0; ct < 4; ++ct)
#pragma unroll
    for (int v = 0; v < 8; ++v)
      sH[rbase + v + half * 8][ct * 16 + m16] = h2n[ct][v];
  __syncthreads();

  for (int i = tid; i < M_ * TD_; i += 128) {
    int r = i / TD_, d = i % TD_;
    float s = sB3[d];
    for (int c = 0; c < H_; ++c) s += sH[r][c] * sW3[c * TD_ + d];
    traj_out[(b * M_ + r) * TD_ + d] = s;
  }
}

// ---------------------------------------------------------------------------
// Kernel 5: TrajScoreSelection — W1 traj-part (60x64) from LDS, WMMA hidden
// layer, dout=1 via shuffles, softmax over the 50 candidates.
// ---------------------------------------------------------------------------
__global__ void __launch_bounds__(128)
score_kernel(MlpP sc, const float* __restrict__ featWS,
             const float* __restrict__ traj, float* __restrict__ score_out)
{
  const int b = blockIdx.x;
  __shared__ __align__(16) _Float16 sW2t[H_ * H_];
  __shared__ __align__(16) _Float16 sA[4][16 * H_];
  __shared__ float sW1t[TD_ * H_];   // W1 rows 128..187 (traj part)
  __shared__ float sFeat[H_], sB1[H_], sG1[H_], sE1[H_], sB2[H_], sG2[H_], sE2[H_];
  __shared__ float sW3[H_];
  __shared__ float sB3v;
  __shared__ float sLog[64];

  const int tid = threadIdx.x;
  for (int i = tid; i < H_ * H_; i += 128) {
    int col = i >> 6, k = i & 63;
    sW2t[i] = (_Float16)sc.W2[k * H_ + col];
  }
  for (int i = tid; i < TD_ * H_; i += 128) sW1t[i] = sc.W1[C_ * H_ + i];
  for (int i = tid; i < H_; i += 128) {
    sFeat[i] = featWS[(3 * B_ + b) * H_ + i];
    sB1[i] = sc.b1[i]; sG1[i] = sc.g1[i]; sE1[i] = sc.e1[i];
    sB2[i] = sc.b2[i]; sG2[i] = sc.g2[i]; sE2[i] = sc.e2[i];
    sW3[i] = sc.W3[i];
  }
  if (tid == 0) sB3v = sc.b3[0];
  __syncthreads();

  const int wave = tid >> 5, lane = tid & 31;
  const int half = lane >> 4, m16 = lane & 15;
  _Float16* aT = sA[wave];
  const int rbase = wave * 16;
  const int row   = rbase + (lane >> 1);
  const int rl    = (row < M_) ? row : (M_ - 1);
  const float* tr = traj + (b * M_ + rl) * TD_;

  v16h bf[4][2];
  load_w2_frags(sW2t, bf);
  float b2v[4], g2v[4], e2v[4], w3v[4];
#pragma unroll
  for (int ct = 0; ct < 4; ++ct) {
    int col = ct * 16 + m16;
    b2v[ct] = sB2[col]; g2v[ct] = sG2[col]; e2v[ct] = sE2[col];
    w3v[ct] = sW3[col];
  }

  {
    float trj[TD_];
    for (int t = 0; t < TD_; ++t) trj[t] = tr[t];
    const int cb = (lane & 1) * 32;
    float h[32]; float s = 0.f, s2 = 0.f;
    for (int j = 0; j < 32; ++j) {
      int c = cb + j;
      float x = sFeat[c] + sB1[c];
      for (int t = 0; t < TD_; ++t) x += trj[t] * sW1t[t * H_ + c];
      h[j] = x; s += x; s2 += x * x;
    }
    s += xr(s, 1); s2 += xr(s2, 1);
    float mu  = s * (1.f / 64.f);
    float var = s2 * (1.f / 64.f) - mu * mu;
    float r   = rsqrtf(var + LN_EPS);
    _Float16* aRow = aT + (lane >> 1) * H_;
    for (int j = 0; j < 32; ++j) {
      int c = cb + j;
      float y = (h[j] - mu) * r * sG1[c] + sE1[c];
      aRow[c] = (_Float16)fmaxf(y, 0.f);
    }
  }
  lds_fence_wave();

  float h2n[4][8];
  hidden_wmma_regs(aT, bf, b2v, g2v, e2v, h2n);

#pragma unroll
  for (int v = 0; v < 8; ++v) {
    float p = 0.f;
#pragma unroll
    for (int ct = 0; ct < 4; ++ct) p += h2n[ct][v] * w3v[ct];
    for (int mk = 1; mk < 16; mk <<= 1) p += xr(p, mk);
    if (m16 == 0) sLog[rbase + v + half * 8] = p + sB3v;
  }
  __syncthreads();

  if (tid == 0) {
    float mx = -INFINITY;
    for (int r = 0; r < M_; ++r) mx = fmaxf(mx, sLog[r]);
    float e[M_]; float ssum = 0.f;
    for (int r = 0; r < M_; ++r) { e[r] = __expf(sLog[r] - mx); ssum += e[r]; }
    float inv = 1.f / ssum;
    for (int r = 0; r < M_; ++r) score_out[b * M_ + r] = e[r] * inv;
  }
}

// ---------------------------------------------------------------------------
extern "C" void kernel_launch(void* const* d_in, const int* in_sizes, int n_in,
                              void* d_out, int out_size, void* d_ws, size_t ws_size,
                              hipStream_t stream)
{
  (void)in_sizes; (void)n_in; (void)out_size; (void)ws_size;

  const float* feat = (const float*)d_in[0];
  const float* cand = (const float*)d_in[1];
  const float* mask = (const float*)d_in[2];

  auto mlp = [&](int base) {
    MlpP p;
    p.W1 = (const float*)d_in[base + 0]; p.b1 = (const float*)d_in[base + 1];
    p.g1 = (const float*)d_in[base + 2]; p.e1 = (const float*)d_in[base + 3];
    p.W2 = (const float*)d_in[base + 4]; p.b2 = (const float*)d_in[base + 5];
    p.g2 = (const float*)d_in[base + 6]; p.e2 = (const float*)d_in[base + 7];
    p.W3 = (const float*)d_in[base + 8]; p.b3 = (const float*)d_in[base + 9];
    return p;
  };
  MlpP prob = mlp(3), offs = mlp(13), mot = mlp(23), scr = mlp(33);

  // Workspace layout (floats): featWS[4*B*H] | logits[B*N] | offsets[B*N*2] | tarSel[B*M*2]
  float* ws      = (float*)d_ws;
  float* featWS  = ws;                         // 65536
  float* logits  = featWS + 4 * B_ * H_;       // 512000
  float* offsets = logits + B_ * N_;           // 1024000
  float* tarSel  = offsets + B_ * N_ * 2;      // 25600   (total ~6.6 MB)

  float* traj  = (float*)d_out;                // [B, M, 60]
  float* score = traj + B_ * M_ * TD_;         // [B, M]

  feat_proj_kernel<<<dim3(B_, 4), H_, 0, stream>>>(feat, prob.W1, offs.W1, mot.W1, scr.W1, featWS);
  stage1_kernel<<<dim3(B_, 2), 128, 0, stream>>>(prob, offs, featWS, cand, mask, logits, offsets);
  topk_kernel<<<B_, 256, 0, stream>>>(logits, cand, offsets, tarSel);
  motion_kernel<<<B_, 128, 0, stream>>>(mot, featWS, tarSel, traj);
  score_kernel<<<B_, 128, 0, stream>>>(scr, featWS, traj, score);
}
